// DeepseekMoE_27736898797646
// MI455X (gfx1250) — matrix operand order repaired and checked by reading, not compiled, not run on hardware
//
#include <hip/hip_runtime.h>
#include <math.h>

// ---------------- problem constants ----------------
#define BD 1
#define SD 2048
#define DD 1024
#define FD 1408
#define ED 16
#define KK 2
#define NT (BD*SD)      // 2048 tokens
#define NKT (NT*KK)     // 4096 (token,k) pairs
#define CAP 320         // ceil(1.25*4096/16)

typedef __bf16 bf16_t;
typedef __bf16  v16bf __attribute__((ext_vector_type(16)));
typedef float   v8f   __attribute__((ext_vector_type(8)));

// ---------------- 1. router: logits -> softmax -> top2 -> renorm ----------------
__global__ __launch_bounds__(64)
void moe_router_kernel(const float* __restrict__ x, const float* __restrict__ rw,
                       const float* __restrict__ rb,
                       int* __restrict__ idx_flat, float* __restrict__ vals_flat)
{
    __shared__ float red[64 * ED];
    const int n = blockIdx.x;
    const int tid = threadIdx.x;

    float acc[ED];
#pragma unroll
    for (int e = 0; e < ED; ++e) acc[e] = 0.f;

    for (int d = tid; d < DD; d += 64) {
        float xv = x[(size_t)n * DD + d];
        const float* w = rw + (size_t)d * ED;
#pragma unroll
        for (int e = 0; e < ED; ++e) acc[e] += xv * w[e];
    }
#pragma unroll
    for (int e = 0; e < ED; ++e) red[tid * ED + e] = acc[e];
    __syncthreads();

    if (tid == 0) {
        float logit[ED];
        for (int e = 0; e < ED; ++e) {
            float s = rb[e];
            for (int t = 0; t < 64; ++t) s += red[t * ED + e];
            logit[e] = s;
        }
        float mx = logit[0];
        for (int e = 1; e < ED; ++e) mx = fmaxf(mx, logit[e]);
        float p[ED], sum = 0.f;
        for (int e = 0; e < ED; ++e) { p[e] = __expf(logit[e] - mx); sum += p[e]; }
        int i0 = 0; float v0 = p[0];
        for (int e = 1; e < ED; ++e) if (p[e] > v0) { v0 = p[e]; i0 = e; }
        int i1 = -1; float v1 = -1.f;
        for (int e = 0; e < ED; ++e) if (e != i0 && p[e] > v1) { v1 = p[e]; i1 = e; }
        v0 /= sum; v1 /= sum;
        float inv = 1.f / (v0 + v1 + 1e-9f);
        idx_flat[2 * n]     = i0;  vals_flat[2 * n]     = v0 * inv;
        idx_flat[2 * n + 1] = i1;  vals_flat[2 * n + 1] = v1 * inv;
    }
}

// ---------------- 2. init slot_token = -1 ----------------
__global__ void moe_init_slots_kernel(int* __restrict__ slot_token)
{
    int i = blockIdx.x * blockDim.x + threadIdx.x;
    if (i < ED * CAP) slot_token[i] = -1;
}

// ---------------- 3. sequential-rank dispatch scan ----------------
__global__ __launch_bounds__(32)
void moe_scan_kernel(const int* __restrict__ idx_flat, const float* __restrict__ vals_flat,
                     int* __restrict__ dest_slot, float* __restrict__ wcomb,
                     int* __restrict__ slot_token)
{
    __shared__ int se[32];
    __shared__ int counts[ED];
    const int tid = threadIdx.x;
    if (tid < ED) counts[tid] = 0;
    __syncthreads();

    for (int base = 0; base < NKT; base += 32) {
        int id = base + tid;
        int e = idx_flat[id];
        se[tid] = e;
        __syncthreads();
        int rank = 0;
        for (int j = 0; j < tid; ++j) rank += (se[j] == e);
        int before = counts[e];
        __syncthreads();
        if (tid < ED) {
            int c = 0;
            for (int j = 0; j < 32; ++j) c += (se[j] == tid);
            counts[tid] += c;
        }
        __syncthreads();
        int pos = before + rank;
        bool within = (pos < CAP);
        dest_slot[id] = within ? (e * CAP + pos) : -1;
        wcomb[id] = within ? vals_flat[id] : 0.f;
        if (within) slot_token[e * CAP + pos] = id >> 1;
    }
}

// ---------------- packing / fragment helpers ----------------
__device__ inline unsigned pk2(float lo, float hi)
{
    union { __bf16 h[2]; unsigned u; } t;
    t.h[0] = (__bf16)lo; t.h[1] = (__bf16)hi;
    return t.u;
}

union FragU { uint4 q[2]; v16bf v; };

// A tile in LDS: [row 0..63][k 0..31], row-major. Fragment per ISA 16-bit A layout.
__device__ inline v16bf frag_a(const bf16_t* lA, int row, int khalf)
{
    const bf16_t* p = lA + row * 32 + khalf * 8;
    FragU f;
    f.q[0] = *(const uint4*)(p);        // K = khalf*8 + 0..7
    f.q[1] = *(const uint4*)(p + 16);   // K = 16 + khalf*8 + 0..7
    return f.v;
}
// B tile in LDS transposed: [n 0..63][k 0..31]. Fragment: same K order as frag_a.
__device__ inline v16bf frag_b(const bf16_t* lB, int col, int khalf)
{
    const bf16_t* p = lB + col * 32 + khalf * 8;
    FragU f;
    f.q[0] = *(const uint4*)(p);
    f.q[1] = *(const uint4*)(p + 16);
    return f.v;
}

// Load 16 consecutive f32 of one A row, pack to 8x(2xbf16).
__device__ inline void load_pack_row(const float* src, unsigned r[8])
{
    const float4* p = (const float4*)src;
    float4 f0 = p[0], f1 = p[1], f2 = p[2], f3 = p[3];
    r[0] = pk2(f0.x, f0.y); r[1] = pk2(f0.z, f0.w);
    r[2] = pk2(f1.x, f1.y); r[3] = pk2(f1.z, f1.w);
    r[4] = pk2(f2.x, f2.y); r[5] = pk2(f2.z, f2.w);
    r[6] = pk2(f3.x, f3.y); r[7] = pk2(f3.z, f3.w);
}
// Load 16 K-strided f32 of one weight column (lane-coalesced across n), pack.
__device__ inline void load_pack_col(const float* W, size_t ldn, unsigned r[8])
{
    float v[16];
#pragma unroll
    for (int j = 0; j < 16; ++j) v[j] = W[(size_t)j * ldn];
#pragma unroll
    for (int j = 0; j < 8; ++j) r[j] = pk2(v[2 * j], v[2 * j + 1]);
}
__device__ inline void store8(bf16_t* dst, const unsigned r[8])
{
    uint4* q = (uint4*)dst;
    q[0] = make_uint4(r[0], r[1], r[2], r[3]);
    q[1] = make_uint4(r[4], r[5], r[6], r[7]);
}
// fast silu(g)*u : v_rcp_f32-based sigmoid (bf16-GEMM precision regime)
__device__ inline float silu_mul(float g, float u)
{
    return __fdividef(g, 1.f + __expf(-g)) * u;
}

// ---------------- 4. fused gate/up GEMM: Out = silu(A@Wg) * (A@Wu) ----------------
// grid: (Nd/64, M/64, E); block 128 = 4 waves in 2x2; double-buffered LDS.
__global__ __launch_bounds__(128)
void moe_gateup_kernel(const float* __restrict__ X, const int* __restrict__ gather,
                       const float* __restrict__ Wg, const float* __restrict__ Wu,
                       float* __restrict__ Out, int M, int Kd, int Nd)
{
    const int e = blockIdx.z;
    const int m_base = blockIdx.y * 64;
    const int n_base = blockIdx.x * 64;
    const int tid = threadIdx.x;
    const int wave = tid >> 5, lane = tid & 31;
    const int wave_m = wave >> 1, wave_n = wave & 1;
    const int khalf = lane >> 4, l16 = lane & 15;

    __shared__ bf16_t lA[2][64 * 32];
    __shared__ bf16_t lBg[2][64 * 32];
    __shared__ bf16_t lBu[2][64 * 32];

    v8f accg[2][2] = {}, accu[2][2] = {};

    const int arow  = tid >> 1;
    const int acol0 = (tid & 1) * 16;
    const int grow  = m_base + arow;
    const int gtok  = gather ? gather[(size_t)e * M + grow] : grow;
    const int bn  = tid & 63;
    const int bkh = tid >> 6;            // k half (16 k's)
    const size_t wbase = (size_t)e * Kd * Nd;
    const float* gB = Wg + wbase + (size_t)bkh * 16 * Nd + n_base + bn;
    const float* uB = Wu + wbase + (size_t)bkh * 16 * Nd + n_base + bn;

    unsigned rA[8], rG[8], rU[8];
    const int T = Kd / 32;

    if (gtok >= 0) load_pack_row(X + (size_t)gtok * Kd + acol0, rA);
    else { for (int i = 0; i < 8; ++i) rA[i] = 0u; }
    load_pack_col(gB, Nd, rG);
    load_pack_col(uB, Nd, rU);
    store8(&lA[0][arow * 32 + acol0], rA);
    store8(&lBg[0][bn * 32 + bkh * 16], rG);
    store8(&lBu[0][bn * 32 + bkh * 16], rU);
    __syncthreads();

    for (int t = 0; t < T; ++t) {
        const int cur = t & 1;
        const bool more = (t + 1 < T);
        if (more) {
            int k0 = (t + 1) * 32;
            if (gtok >= 0) load_pack_row(X + (size_t)gtok * Kd + k0 + acol0, rA);
            load_pack_col(gB + (size_t)k0 * Nd, Nd, rG);
            load_pack_col(uB + (size_t)k0 * Nd, Nd, rU);
            if (t + 2 < T) {   // prefetch hint for tile after next (global_prefetch_b8)
                size_t kp = (size_t)(t + 2) * 32 * Nd;
                __builtin_prefetch(gB + kp, 0, 1);
                __builtin_prefetch(uB + kp, 0, 1);
            }
        }

        v16bf a0 = frag_a(lA[cur], wave_m * 32 + l16, khalf);
        v16bf a1 = frag_a(lA[cur], wave_m * 32 + 16 + l16, khalf);
#pragma unroll
        for (int s = 0; s < 2; ++s) {
            v16bf bg = frag_b(lBg[cur], wave_n * 32 + s * 16 + l16, khalf);
            v16bf bu = frag_b(lBu[cur], wave_n * 32 + s * 16 + l16, khalf);
            accg[0][s] = __builtin_amdgcn_wmma_f32_16x16x32_bf16(false, a0, false, bg,
                                                                 (short)0, accg[0][s], false, false);
            accg[1][s] = __builtin_amdgcn_wmma_f32_16x16x32_bf16(false, a1, false, bg,
                                                                 (short)0, accg[1][s], false, false);
            accu[0][s] = __builtin_amdgcn_wmma_f32_16x16x32_bf16(false, a0, false, bu,
                                                                 (short)0, accu[0][s], false, false);
            accu[1][s] = __builtin_amdgcn_wmma_f32_16x16x32_bf16(false, a1, false, bu,
                                                                 (short)0, accu[1][s], false, false);
        }

        if (more) {
            const int nxt = cur ^ 1;
            store8(&lA[nxt][arow * 32 + acol0], rA);
            store8(&lBg[nxt][bn * 32 + bkh * 16], rG);
            store8(&lBu[nxt][bn * 32 + bkh * 16], rU);
        }
        __syncthreads();
    }

#pragma unroll
    for (int sub = 0; sub < 2; ++sub) {
        const int mrow0 = m_base + wave_m * 32 + sub * 16 + khalf * 8;
#pragma unroll
        for (int s = 0; s < 2; ++s) {
            const int n = n_base + wave_n * 32 + s * 16 + l16;
#pragma unroll
            for (int r = 0; r < 8; ++r)
                Out[((size_t)e * M + mrow0 + r) * Nd + n] =
                    silu_mul(accg[sub][s][r], accu[sub][s][r]);
        }
    }
}

// ---------------- 5. down GEMM: Out = A @ Wd ----------------
__global__ __launch_bounds__(128)
void moe_down_kernel(const float* __restrict__ A, const float* __restrict__ Wd,
                     float* __restrict__ Out, int M, int Kd, int Nd)
{
    const int e = blockIdx.z;
    const int m_base = blockIdx.y * 64;
    const int n_base = blockIdx.x * 64;
    const int tid = threadIdx.x;
    const int wave = tid >> 5, lane = tid & 31;
    const int wave_m = wave >> 1, wave_n = wave & 1;
    const int khalf = lane >> 4, l16 = lane & 15;

    __shared__ bf16_t lA[2][64 * 32];
    __shared__ bf16_t lB[2][64 * 32];

    v8f acc[2][2] = {};

    const int arow  = tid >> 1;
    const int acol0 = (tid & 1) * 16;
    const int bn  = tid & 63;
    const int bkh = tid >> 6;
    const size_t wbase = (size_t)e * Kd * Nd;
    const float* gA = A + ((size_t)e * M + m_base + arow) * Kd + acol0;
    const float* gB = Wd + wbase + (size_t)bkh * 16 * Nd + n_base + bn;

    unsigned rA[8], rB[8];
    const int T = Kd / 32;

    load_pack_row(gA, rA);
    load_pack_col(gB, Nd, rB);
    store8(&lA[0][arow * 32 + acol0], rA);
    store8(&lB[0][bn * 32 + bkh * 16], rB);
    __syncthreads();

    for (int t = 0; t < T; ++t) {
        const int cur = t & 1;
        const bool more = (t + 1 < T);
        if (more) {
            int k0 = (t + 1) * 32;
            load_pack_row(gA + k0, rA);
            load_pack_col(gB + (size_t)k0 * Nd, Nd, rB);
            if (t + 2 < T) {
                __builtin_prefetch(gB + (size_t)(t + 2) * 32 * Nd, 0, 1);
            }
        }

        v16bf a0 = frag_a(lA[cur], wave_m * 32 + l16, khalf);
        v16bf a1 = frag_a(lA[cur], wave_m * 32 + 16 + l16, khalf);
#pragma unroll
        for (int s = 0; s < 2; ++s) {
            v16bf b = frag_b(lB[cur], wave_n * 32 + s * 16 + l16, khalf);
            acc[0][s] = __builtin_amdgcn_wmma_f32_16x16x32_bf16(false, a0, false, b,
                                                                (short)0, acc[0][s], false, false);
            acc[1][s] = __builtin_amdgcn_wmma_f32_16x16x32_bf16(false, a1, false, b,
                                                                (short)0, acc[1][s], false, false);
        }

        if (more) {
            const int nxt = cur ^ 1;
            store8(&lA[nxt][arow * 32 + acol0], rA);
            store8(&lB[nxt][bn * 32 + bkh * 16], rB);
        }
        __syncthreads();
    }

#pragma unroll
    for (int sub = 0; sub < 2; ++sub) {
        const int mrow0 = m_base + wave_m * 32 + sub * 16 + khalf * 8;
#pragma unroll
        for (int s = 0; s < 2; ++s) {
            const int n = n_base + wave_n * 32 + s * 16 + l16;
#pragma unroll
            for (int r = 0; r < 8; ++r)
                Out[((size_t)e * M + mrow0 + r) * Nd + n] = acc[sub][s][r];
        }
    }
}

// ---------------- 6. combine: out = x + shared + sum_k w*out_slot[slot] ----------------
__global__ __launch_bounds__(256)
void moe_combine_kernel(const float* __restrict__ x, const float* __restrict__ shared_out,
                        const float* __restrict__ out_slot,
                        const int* __restrict__ dest_slot, const float* __restrict__ wcomb,
                        float* __restrict__ out)
{
    const int n = blockIdx.y;
    const int d = blockIdx.x * blockDim.x + threadIdx.x;
    float r = x[(size_t)n * DD + d] + shared_out[(size_t)n * DD + d];
    int s0 = dest_slot[2 * n], s1 = dest_slot[2 * n + 1];
    float w0 = wcomb[2 * n], w1 = wcomb[2 * n + 1];
    if (s0 >= 0) r += w0 * out_slot[(size_t)s0 * DD + d];
    if (s1 >= 0) r += w1 * out_slot[(size_t)s1 * DD + d];
    out[(size_t)n * DD + d] = r;
}

// ---------------- host launcher ----------------
extern "C" void kernel_launch(void* const* d_in, const int* in_sizes, int n_in,
                              void* d_out, int out_size, void* d_ws, size_t ws_size,
                              hipStream_t stream)
{
    const float* x   = (const float*)d_in[0];
    const float* rw  = (const float*)d_in[1];
    const float* rb  = (const float*)d_in[2];
    const float* wg  = (const float*)d_in[3];
    const float* wu  = (const float*)d_in[4];
    const float* wd  = (const float*)d_in[5];
    const float* swg = (const float*)d_in[6];
    const float* swu = (const float*)d_in[7];
    const float* swd = (const float*)d_in[8];
    float* out = (float*)d_out;

    char* ws = (char*)d_ws;
    size_t o = 0;
    auto alloc = [&](size_t bytes) -> void* {
        o = (o + 255) & ~(size_t)255;
        void* p = ws + o;
        o += bytes;
        return p;
    };
    int*   idx_flat   = (int*)  alloc((size_t)NKT * 4);
    float* vals_flat  = (float*)alloc((size_t)NKT * 4);
    int*   dest_slot  = (int*)  alloc((size_t)NKT * 4);
    float* wcomb      = (float*)alloc((size_t)NKT * 4);
    int*   slot_token = (int*)  alloc((size_t)ED * CAP * 4);
    float* act        = (float*)alloc((size_t)ED * CAP * FD * 4);
    float* out_slot   = (float*)alloc((size_t)ED * CAP * DD * 4);
    float* h1         = (float*)alloc((size_t)NT * FD * 4);
    float* sh_out     = (float*)alloc((size_t)NT * DD * 4);
    (void)ws_size; (void)in_sizes; (void)n_in; (void)out_size;

    moe_router_kernel<<<dim3(NT), dim3(64), 0, stream>>>(x, rw, rb, idx_flat, vals_flat);
    moe_init_slots_kernel<<<dim3((ED * CAP + 255) / 256), dim3(256), 0, stream>>>(slot_token);
    moe_scan_kernel<<<dim3(1), dim3(32), 0, stream>>>(idx_flat, vals_flat,
                                                      dest_slot, wcomb, slot_token);
    moe_gateup_kernel<<<dim3(FD / 64, CAP / 64, ED), dim3(128), 0, stream>>>(
        x, slot_token, wg, wu, act, CAP, DD, FD);
    moe_gateup_kernel<<<dim3(FD / 64, NT / 64, 1), dim3(128), 0, stream>>>(
        x, nullptr, swg, swu, h1, NT, DD, FD);
    moe_down_kernel<<<dim3(DD / 64, CAP / 64, ED), dim3(128), 0, stream>>>(
        act, wd, out_slot, CAP, FD, DD);
    moe_down_kernel<<<dim3(DD / 64, NT / 64, 1), dim3(128), 0, stream>>>(
        h1, swd, sh_out, NT, FD, DD);
    moe_combine_kernel<<<dim3(DD / 256, NT), dim3(256), 0, stream>>>(
        x, sh_out, out_slot, dest_slot, wcomb, out);
}
